// MoETransformerBlock_11639361372757
// MI455X (gfx1250) — matrix operand order repaired and checked by reading, not compiled, not run on hardware
//
#include <hip/hip_runtime.h>
#include <math.h>

// ---------------------------------------------------------------------------
// MoE Transformer block for MI455X (gfx1250, wave32, WMMA).
// bf16 operands + f32 WMMA accumulation for all matmuls
// (v_wmma_f32_16x16x32_bf16); f32 VALU for LN/softmax/GELU/router.
// Fragment-major LDS staging so every WMMA operand is 2x ds_load_b128.
// Softmax row-sums computed as P . ones via WMMA (no cross-lane reduce);
// 1/sqrt(DH) folded into the Q-projection epilogue.
// ---------------------------------------------------------------------------

typedef __attribute__((ext_vector_type(16))) __bf16 v16bf;
typedef __attribute__((ext_vector_type(8)))  float  v8f;

static constexpr int Bc   = 4;
static constexpr int Sc   = 1024;
static constexpr int Ec   = 1024;
static constexpr int Hc   = 16;
static constexpr int NEc  = 8;
static constexpr int DFFc = 4096;
static constexpr int DHc  = 64;
static constexpr int NTOK = Bc * Sc;   // 4096

// ---------------- small helpers ----------------
__device__ inline __bf16 f2bf(float x) {
  union { float f; unsigned u; } v; v.f = x;
  unsigned r = v.u + 0x7FFFu + ((v.u >> 16) & 1u);   // round-to-nearest-even
  unsigned short h = (unsigned short)(r >> 16);
  __bf16 o; __builtin_memcpy(&o, &h, 2); return o;
}

__device__ inline float gelu_exact(float x) {
  return 0.5f * x * (1.f + erff(x * 0.70710678118654752f));
}

// A fragment (16x32 bf16, MxK) from row-major LDS tile with 16B-aligned rows:
// lanes 0-15: M=lane, K={0..7,16..23}; lanes 16-31: M=lane-16, K={8..15,24..31}
// ld must make (m*ld + kb)*2 a multiple of 16 (ld = 48 works).
__device__ inline v16bf frag_a_lds(const __bf16* base, int ld, int lane) {
  const int m  = lane & 15;
  const int kb = (lane >> 4) << 3;
  const __bf16* p = base + m * ld + kb;
  union { uint4 u[2]; v16bf v; } t;
  t.u[0] = *reinterpret_cast<const uint4*>(p);       // ds_load_b128
  t.u[1] = *reinterpret_cast<const uint4*>(p + 16);  // ds_load_b128
  return t.v;
}

// B fragment from fragment-major LDS chunk: [32 lanes][16 contiguous values]
__device__ inline v16bf frag_packed(const __bf16* chunk, int lane) {
  const uint4* p = reinterpret_cast<const uint4*>(chunk + lane * 16);
  union { uint4 u[2]; v16bf v; } t;
  t.u[0] = p[0];   // ds_load_b128
  t.u[1] = p[1];   // ds_load_b128
  return t.v;
}

// ---------------- f32 -> bf16 convert ----------------
__global__ __launch_bounds__(256) void cvt_f32_bf16(const float* __restrict__ in,
                                                    __bf16* __restrict__ out,
                                                    size_t n4) {
  size_t i = (size_t)blockIdx.x * blockDim.x + threadIdx.x;
  const size_t stride = (size_t)gridDim.x * blockDim.x;
  for (; i < n4; i += stride) {
    const float4 v = reinterpret_cast<const float4*>(in)[i];
    union { __bf16 b[4]; uint2 u; } t;
    t.b[0] = f2bf(v.x); t.b[1] = f2bf(v.y); t.b[2] = f2bf(v.z); t.b[3] = f2bf(v.w);
    reinterpret_cast<uint2*>(out)[i] = t.u;
  }
}

// ---------------- LayerNorm (one block per token row) ----------------
__global__ __launch_bounds__(256) void ln_kernel(const float* __restrict__ x,
                                                 const float* __restrict__ g,
                                                 const float* __restrict__ bb,
                                                 float* __restrict__ outf,
                                                 __bf16* __restrict__ outb) {
  const int row = blockIdx.x;
  const int tid = threadIdx.x;
  const float* xr = x + (size_t)row * Ec;
  float v[4]; float s = 0.f, s2 = 0.f;
#pragma unroll
  for (int i = 0; i < 4; ++i) {
    v[i] = xr[tid + i * 256]; s += v[i]; s2 += v[i] * v[i];
  }
#pragma unroll
  for (int msk = 1; msk < 32; msk <<= 1) {
    s  += __shfl_xor(s,  msk, 32);
    s2 += __shfl_xor(s2, msk, 32);
  }
  __shared__ float rs[8], rs2[8];
  __shared__ float mean_s, rstd_s;
  const int lane = tid & 31, w = tid >> 5;
  if (lane == 0) { rs[w] = s; rs2[w] = s2; }
  __syncthreads();
  if (tid == 0) {
    float a = 0.f, b2 = 0.f;
#pragma unroll
    for (int i = 0; i < 8; ++i) { a += rs[i]; b2 += rs2[i]; }
    const float mean = a * (1.f / Ec);
    const float var  = b2 * (1.f / Ec) - mean * mean;
    mean_s = mean; rstd_s = rsqrtf(var + 1e-5f);
  }
  __syncthreads();
  const float mean = mean_s, rstd = rstd_s;
#pragma unroll
  for (int i = 0; i < 4; ++i) {
    const int c = tid + i * 256;
    const float y = (v[i] - mean) * rstd * g[c] + bb[c];
    if (outf) outf[(size_t)row * Ec + c] = y;
    outb[(size_t)row * Ec + c] = f2bf(y);
  }
}

// ---------------- bf16 WMMA GEMM: C[M,N] = A[M,K] @ B[K,N] (+epilogue) -----
static constexpr int BM = 128, BN = 64, BK = 32;
static constexpr int LDA_T = 48;   // A row pitch (96B -> 16B-aligned frags)

template <bool GELU_, bool ADDSRC, bool ACCUM, bool ROWSCALE, bool QKV,
          bool OUTF, bool OUTB>
__global__ __launch_bounds__(256) void gemm_bf16_wmma(
    const __bf16* __restrict__ A, const __bf16* __restrict__ Bw,
    const float* __restrict__ bias, const float* __restrict__ addsrc,
    const float* __restrict__ rowgate, float* __restrict__ outf,
    __bf16* __restrict__ outb, int M, int N, int K, float oscale) {
  (void)M;
  __shared__ __align__(16) __bf16 Asm[BM * LDA_T];     // 12 KB, row-major
  __shared__ __align__(16) __bf16 BsmF[4 * 32 * 16];   // 4 KB, fragment-major
  const int tid  = threadIdx.x;
  const int lane = tid & 31;
  const int w    = tid >> 5;          // 8 waves
  const int wm   = w >> 1;            // 0..3  (32-row strip)
  const int wn   = w & 1;             // 0..1  (32-col strip)
  const int m0   = blockIdx.y * BM;
  const int n0   = blockIdx.x * BN;

  v8f acc[2][2] = {};
  uint2 ra[4], rb[2];

  // preload first K-tile into registers
#pragma unroll
  for (int r = 0; r < 4; ++r) {
    const int idx = (tid + r * 256) * 4;
    ra[r] = *reinterpret_cast<const uint2*>(
        &A[(size_t)(m0 + (idx >> 5)) * K + (idx & 31)]);
  }
#pragma unroll
  for (int r = 0; r < 2; ++r) {
    const int idx = (tid + r * 256) * 4;
    rb[r] = *reinterpret_cast<const uint2*>(
        &Bw[(size_t)(idx >> 6) * N + n0 + (idx & 63)]);
  }

  for (int k0 = 0; k0 < K; k0 += BK) {
    __syncthreads();   // previous tile fully consumed
    // commit staged registers to LDS
#pragma unroll
    for (int r = 0; r < 4; ++r) {
      const int idx = (tid + r * 256) * 4;
      *reinterpret_cast<uint2*>(&Asm[(idx >> 5) * LDA_T + (idx & 31)]) = ra[r];
    }
#pragma unroll
    for (int r = 0; r < 2; ++r) {
      const int idx = (tid + r * 256) * 4;
      const int k = idx >> 6, col = idx & 63;
      union { uint2 u; __bf16 b[4]; } t; t.u = rb[r];
      const int pos = (k & 7) | ((k >> 4) << 3), khi = ((k >> 3) & 1) << 4;
#pragma unroll
      for (int q = 0; q < 4; ++q) {
        const int n = col + q;   // chunk = n>>4, lane = (n&15)|khi, slot = pos
        BsmF[((((n >> 4) << 5) + (n & 15) + khi) << 4) + pos] = t.b[q];
      }
    }
    __syncthreads();
    // stage next tile (global latency overlaps the WMMA burst below)
    if (k0 + BK < K) {
      const int kn = k0 + BK;
#pragma unroll
      for (int r = 0; r < 4; ++r) {
        const int idx = (tid + r * 256) * 4;
        ra[r] = *reinterpret_cast<const uint2*>(
            &A[(size_t)(m0 + (idx >> 5)) * K + kn + (idx & 31)]);
      }
#pragma unroll
      for (int r = 0; r < 2; ++r) {
        const int idx = (tid + r * 256) * 4;
        rb[r] = *reinterpret_cast<const uint2*>(
            &Bw[(size_t)(kn + (idx >> 6)) * N + n0 + (idx & 63)]);
      }
      if (k0 + 2 * BK < K)   // warm L2 for the tile after next
        __builtin_prefetch(&A[(size_t)(m0 + (tid >> 1)) * K + k0 + 2 * BK], 0, 1);
    }

    v16bf af[2], bfr[2];
    af[0]  = frag_a_lds(&Asm[(wm * 32) * LDA_T], LDA_T, lane);
    af[1]  = frag_a_lds(&Asm[(wm * 32 + 16) * LDA_T], LDA_T, lane);
    bfr[0] = frag_packed(&BsmF[(wn * 2 + 0) << 9], lane);
    bfr[1] = frag_packed(&BsmF[(wn * 2 + 1) << 9], lane);
#pragma unroll
    for (int i = 0; i < 2; ++i)
#pragma unroll
      for (int j = 0; j < 2; ++j)
        acc[i][j] = __builtin_amdgcn_wmma_f32_16x16x32_bf16(
            false, af[i], false, bfr[j], (short)0, acc[i][j], false, false);
  }

  // epilogue; C-frag layout: lanes 0-15 rows mb+0..7 (vgpr r), col = lane&15
  const int nlane = lane & 15;
  const int mb    = (lane >> 4) << 3;
#pragma unroll
  for (int i = 0; i < 2; ++i) {
#pragma unroll
    for (int j = 0; j < 2; ++j) {
      const int ncol = n0 + wn * 32 + j * 16 + nlane;
#pragma unroll
      for (int r = 0; r < 8; ++r) {
        const int mrow = m0 + wm * 32 + i * 16 + mb + r;
        float v = (acc[i][j][r] + bias[ncol]) * oscale;
        if (GELU_)    v = gelu_exact(v);
        if (ROWSCALE) v *= rowgate[(size_t)mrow * NEc];
        float o = v;
        if (ADDSRC) o += addsrc[(size_t)mrow * N + ncol];
        if (ACCUM)  o += outf[(size_t)mrow * N + ncol];
        if (OUTF) outf[(size_t)mrow * N + ncol] = o;
        if (OUTB) {
          if (QKV) {
            const int bb_ = mrow >> 10, ss = mrow & (Sc - 1);
            const int hh  = ncol >> 6,  dd = ncol & (DHc - 1);
            outb[((((size_t)bb_ * Hc + hh) * Sc + ss) << 6) + dd] = f2bf(o);
          } else {
            outb[(size_t)mrow * N + ncol] = f2bf(o);
          }
        }
      }
    }
  }
}

// ---------------- flash attention (bf16 WMMA, f32 online softmax) ----------
static constexpr int QT = 128;   // q rows per block (16 per wave)
static constexpr int KT = 32;    // kv chunk
static constexpr int LP = 48;    // P staging row pitch (16B-aligned frags)

__global__ __launch_bounds__(256) void attn_kernel(const __bf16* __restrict__ qg,
                                                   const __bf16* __restrict__ kg,
                                                   const __bf16* __restrict__ vg,
                                                   __bf16* __restrict__ ctx) {
  // K^T tile, fragment-major: chunk (c over dh-32, j over kv-16) -> [32][16]
  __shared__ __align__(16) __bf16 KtF[4 * 32 * 16];   // 4 KB
  // V tile, fragment-major: chunk jn over d-16 -> [32][16]
  __shared__ __align__(16) __bf16 VtF[4 * 32 * 16];   // 4 KB
  __shared__ __align__(16) __bf16 Pt[8 * 16 * LP];    // 12 KB, row-major/wave
  const int tid = threadIdx.x, lane = tid & 31, w = tid >> 5;
  const int q0 = blockIdx.x * QT;
  const int h  = blockIdx.y, b = blockIdx.z;
  const size_t base = (((size_t)b * Hc + h) * Sc) * DHc;

  // constant all-ones B fragment (for rowsum = P . ones via WMMA)
  v16bf onesb;
#pragma unroll
  for (int i = 0; i < 16; ++i) onesb[i] = f2bf(1.0f);

  // Q fragments (already scaled by 1/sqrt(DH) in the projection epilogue)
  const int qrow = q0 + w * 16;
  v16bf qf[2];
  {
    const int m  = lane & 15;
    const int kb = (lane >> 4) << 4;
#pragma unroll
    for (int c = 0; c < 2; ++c) {
      const __bf16* qp = qg + base + (size_t)(qrow + m) * DHc + c * 32 + kb;
      union { uint4 u[2]; v16bf v; } t;
      t.u[0] = *reinterpret_cast<const uint4*>(qp);
      t.u[1] = *reinterpret_cast<const uint4*>(qp + 8);
      qf[c] = t.v;
    }
  }

  float mrow[8], lrow[8];
#pragma unroll
  for (int r = 0; r < 8; ++r) { mrow[r] = -1e30f; lrow[r] = 0.f; }
  v8f o[4] = {};

  for (int kv0 = 0; kv0 < Sc; kv0 += KT) {
    // K chunk -> KtF: element (kv,d): chunk c=d>>5, j=kv>>4,
    //   lane l=(kv&15)|(((d>>4)&1)<<4), slot = d&15 (4 consecutive d -> uint2)
#pragma unroll
    for (int r = 0; r < 2; ++r) {
      const int idx = (tid + r * 256) * 4;
      const int kv = idx >> 6, d = idx & 63;
      const uint2 u = *reinterpret_cast<const uint2*>(
          &kg[base + (size_t)(kv0 + kv) * DHc + d]);
      const int c = d >> 5, j = kv >> 4;
      const int l = (kv & 15) | (((d >> 4) & 1) << 4);
      *reinterpret_cast<uint2*>(
          &KtF[((((c << 1) + j) << 5) + l) * 16 + (d & 15)]) = u;
    }
    // V chunk -> VtF: element (kv,d): chunk jn=d>>4, lane=(d&15)|(((kv>>3)&1)<<4),
    //   slot = (kv&7)|((kv>>4)<<3) (scatter b16 stores)
#pragma unroll
    for (int r = 0; r < 2; ++r) {
      const int idx = (tid + r * 256) * 4;
      const int kv = idx >> 6, d = idx & 63;
      union { uint2 u; __bf16 bv[4]; } t;
      t.u = *reinterpret_cast<const uint2*>(
          &vg[base + (size_t)(kv0 + kv) * DHc + d]);
      const int khi = ((kv >> 3) & 1) << 4, pos = (kv & 7) | ((kv >> 4) << 3);
#pragma unroll
      for (int q = 0; q < 4; ++q) {
        const int dd = d + q;
        VtF[((((dd >> 4) << 5) + (dd & 15) + khi) << 4) + pos] = t.bv[q];
      }
    }
    __syncthreads();

    // scores: S(16x32) = Q(16x64) . K^T   (pre-scaled Q)
    v8f sa[2] = {};
#pragma unroll
    for (int j = 0; j < 2; ++j)
#pragma unroll
      for (int c = 0; c < 2; ++c)
        sa[j] = __builtin_amdgcn_wmma_f32_16x16x32_bf16(
            false, qf[c], false,
            frag_packed(&KtF[((c << 1) + j) << 9], lane),
            (short)0, sa[j], false, false);

    // row-max via shfl butterfly (16-lane halves hold complete rows), then exp
    float pm[2][8], corr[8];
#pragma unroll
    for (int r = 0; r < 8; ++r) {
      float mx = fmaxf(sa[0][r], sa[1][r]);
#pragma unroll
      for (int msk = 1; msk < 16; msk <<= 1) mx = fmaxf(mx, __shfl_xor(mx, msk, 32));
      const float nm = fmaxf(mrow[r], mx);
      pm[0][r] = __expf(sa[0][r] - nm);
      pm[1][r] = __expf(sa[1][r] - nm);
      corr[r]  = __expf(mrow[r] - nm);
      mrow[r]  = nm;
    }
#pragma unroll
    for (int jn = 0; jn < 4; ++jn)
#pragma unroll
      for (int r = 0; r < 8; ++r) o[jn][r] *= corr[r];

    // stage P (C-frag -> A-frag reshape through per-wave LDS region)
    {
      const int n = lane & 15, mb = (lane >> 4) << 3;
      __bf16* pp = &Pt[w * 16 * LP];
#pragma unroll
      for (int j = 0; j < 2; ++j)
#pragma unroll
        for (int r = 0; r < 8; ++r)
          pp[(mb + r) * LP + j * 16 + n] = f2bf(pm[j][r]);
    }
    __syncthreads();

    const v16bf pf = frag_a_lds(&Pt[w * 16 * LP], LP, lane);

    // row-sums in one WMMA: sums(16x16) = P(16x32) . ones(32x16)
    v8f sums = __builtin_amdgcn_wmma_f32_16x16x32_bf16(
        false, pf, false, onesb, (short)0, v8f{}, false, false);
#pragma unroll
    for (int r = 0; r < 8; ++r) lrow[r] = lrow[r] * corr[r] + sums[r];

    // O(16x64) += P(16x32) . V(32x64)
#pragma unroll
    for (int jn = 0; jn < 4; ++jn)
      o[jn] = __builtin_amdgcn_wmma_f32_16x16x32_bf16(
          false, pf, false, frag_packed(&VtF[jn << 9], lane),
          (short)0, o[jn], false, false);
    __syncthreads();
  }

  // write ctx[b, s, h*64 + d] (bf16)
  {
    const int n = lane & 15, mb = (lane >> 4) << 3;
#pragma unroll
    for (int r = 0; r < 8; ++r) {
      const float inv = 1.f / lrow[r];
      const int srow = q0 + w * 16 + mb + r;
#pragma unroll
      for (int jn = 0; jn < 4; ++jn) {
        const int d = jn * 16 + n;
        ctx[((size_t)(b * Sc + srow)) * Ec + h * DHc + d] = f2bf(o[jn][r] * inv);
      }
    }
  }
}

// ---------------- router: logits, top-2 gates, aux-loss accumulators -------
__global__ __launch_bounds__(256) void router_kernel(const float* __restrict__ h2,
                                                     const float* __restrict__ wg,
                                                     float* __restrict__ gates,
                                                     float* __restrict__ psum,
                                                     float* __restrict__ fsum) {
  const int row = blockIdx.x;
  const int tid = threadIdx.x, lane = tid & 31, w = tid >> 5;  // wave = expert
  const float* hr = h2 + (size_t)row * Ec;
  float acc = 0.f;
  for (int e = lane; e < Ec; e += 32) acc += hr[e] * wg[(size_t)e * NEc + w];
#pragma unroll
  for (int msk = 1; msk < 32; msk <<= 1) acc += __shfl_xor(acc, msk, 32);
  __shared__ float lg[NEc];
  if (lane == 0) lg[w] = acc;
  __syncthreads();
  if (tid == 0) {
    float l[NEc];
#pragma unroll
    for (int n = 0; n < NEc; ++n) l[n] = lg[n];
    int i0 = 0;
    for (int n = 1; n < NEc; ++n) if (l[n] > l[i0]) i0 = n;
    int i1 = -1;
    for (int n = 0; n < NEc; ++n) {
      if (n == i0) continue;
      if (i1 < 0 || l[n] > l[i1]) i1 = n;
    }
    const float m2 = fmaxf(l[i0], l[i1]);
    const float e0 = expf(l[i0] - m2), e1 = expf(l[i1] - m2);
    const float inv = 1.f / (e0 + e1);
    const float g0 = e0 * inv, g1 = e1 * inv;
    float mx = l[0];
    for (int n = 1; n < NEc; ++n) mx = fmaxf(mx, l[n]);
    float den = 0.f, pr[NEc];
    for (int n = 0; n < NEc; ++n) { pr[n] = expf(l[n] - mx); den += pr[n]; }
    const float rden = 1.f / den;
    for (int n = 0; n < NEc; ++n) {
      const float g = (n == i0) ? g0 : (n == i1) ? g1 : 0.f;
      gates[(size_t)row * NEc + n] = g;
      atomicAdd(&psum[n], pr[n] * rden);
      if (g > 0.f) atomicAdd(&fsum[n], 1.f);
    }
  }
}

__global__ void zero_stats(float* p) {
  if (threadIdx.x < 16) p[threadIdx.x] = 0.f;
}

__global__ void aux_kernel(const float* __restrict__ psum,
                           const float* __restrict__ fsum,
                           float* __restrict__ out) {
  if (threadIdx.x == 0 && blockIdx.x == 0) {
    float a = 0.f;
    for (int n = 0; n < NEc; ++n)
      a += (psum[n] / (float)NTOK) * (fsum[n] / (float)NTOK);
    out[0] = (float)NEc * a;
  }
}

// ---------------------------------------------------------------------------
extern "C" void kernel_launch(void* const* d_in, const int* in_sizes, int n_in,
                              void* d_out, int out_size, void* d_ws, size_t ws_size,
                              hipStream_t stream) {
  (void)in_sizes; (void)n_in; (void)out_size; (void)ws_size;
  const float* x     = (const float*)d_in[0];
  const float* ln1_g = (const float*)d_in[1];
  const float* ln1_b = (const float*)d_in[2];
  const float* wq    = (const float*)d_in[3];
  const float* bq    = (const float*)d_in[4];
  const float* wk    = (const float*)d_in[5];
  const float* bk    = (const float*)d_in[6];
  const float* wv    = (const float*)d_in[7];
  const float* bv    = (const float*)d_in[8];
  const float* wo    = (const float*)d_in[9];
  const float* bo    = (const float*)d_in[10];
  const float* ln2_g = (const float*)d_in[11];
  const float* ln2_b = (const float*)d_in[12];
  const float* sw1   = (const float*)d_in[13];
  const float* sb1   = (const float*)d_in[14];
  const float* sw2   = (const float*)d_in[15];
  const float* sb2   = (const float*)d_in[16];
  const float* ew1   = (const float*)d_in[17];
  const float* eb1   = (const float*)d_in[18];
  const float* ew2   = (const float*)d_in[19];
  const float* eb2   = (const float*)d_in[20];
  const float* wgate = (const float*)d_in[21];
  float* outf = (float*)d_out;

  // workspace carve-up (256B aligned)
  char* ws = (char*)d_ws;
  size_t off = 0;
  auto alloc = [&](size_t bytes) -> void* {
    void* p = ws + off;
    off += (bytes + 255) & ~(size_t)255;
    return p;
  };
  const size_t NE_ = (size_t)NTOK * Ec;
  const size_t NDF = (size_t)NTOK * DFFc;
  const size_t WEE = (size_t)Ec * Ec;
  const size_t WED = (size_t)Ec * DFFc;
  __bf16* hb    = (__bf16*)alloc(NE_ * 2);
  __bf16* qb    = (__bf16*)alloc(NE_ * 2);
  __bf16* kb    = (__bf16*)alloc(NE_ * 2);
  __bf16* vb    = (__bf16*)alloc(NE_ * 2);
  __bf16* ctxb  = (__bf16*)alloc(NE_ * 2);
  float*  x1    = (float*)alloc(NE_ * 4);
  float*  h2    = (float*)alloc(NE_ * 4);
  __bf16* h2b   = (__bf16*)alloc(NE_ * 2);
  __bf16* a1b   = (__bf16*)alloc(NDF * 2);
  __bf16* eact  = (__bf16*)alloc(NDF * 2);
  float*  gates = (float*)alloc((size_t)NTOK * NEc * 4);
  float*  stats = (float*)alloc(64);          // psum[8] + fsum[8]
  __bf16* wqb   = (__bf16*)alloc(WEE * 2);
  __bf16* wkb   = (__bf16*)alloc(WEE * 2);
  __bf16* wvb   = (__bf16*)alloc(WEE * 2);
  __bf16* wob   = (__bf16*)alloc(WEE * 2);
  __bf16* sw1b  = (__bf16*)alloc(WED * 2);
  __bf16* sw2b  = (__bf16*)alloc(WED * 2);
  __bf16* ew1b  = (__bf16*)alloc(WED * 2);
  __bf16* ew2b  = (__bf16*)alloc(WED * 2);

  const dim3 blk(256);
  const dim3 gE (Ec / BN,  NTOK / BM);   // N=1024 GEMMs
  const dim3 gD (DFFc / BN, NTOK / BM);  // N=4096 GEMMs

  // 1) weight converts
  cvt_f32_bf16<<<1024, blk, 0, stream>>>(wq,  wqb,  WEE / 4);
  cvt_f32_bf16<<<1024, blk, 0, stream>>>(wk,  wkb,  WEE / 4);
  cvt_f32_bf16<<<1024, blk, 0, stream>>>(wv,  wvb,  WEE / 4);
  cvt_f32_bf16<<<1024, blk, 0, stream>>>(wo,  wob,  WEE / 4);
  cvt_f32_bf16<<<1024, blk, 0, stream>>>(sw1, sw1b, WED / 4);
  cvt_f32_bf16<<<1024, blk, 0, stream>>>(sw2, sw2b, WED / 4);

  // 2) LN1 -> hb (bf16)
  ln_kernel<<<NTOK, blk, 0, stream>>>(x, ln1_g, ln1_b, nullptr, hb);

  // 3) Q,K,V projections -> [B,H,S,DH] bf16; Q pre-scaled by 1/sqrt(DH)
  gemm_bf16_wmma<false,false,false,false,true,false,true>
      <<<gE, blk, 0, stream>>>(hb, wqb, bq, nullptr, nullptr, nullptr, qb,
                               NTOK, Ec, Ec, 0.125f);
  gemm_bf16_wmma<false,false,false,false,true,false,true>
      <<<gE, blk, 0, stream>>>(hb, wkb, bk, nullptr, nullptr, nullptr, kb,
                               NTOK, Ec, Ec, 1.0f);
  gemm_bf16_wmma<false,false,false,false,true,false,true>
      <<<gE, blk, 0, stream>>>(hb, wvb, bv, nullptr, nullptr, nullptr, vb,
                               NTOK, Ec, Ec, 1.0f);

  // 4) flash attention -> ctxb [B,S,E]
  attn_kernel<<<dim3(Sc / QT, Hc, Bc), blk, 0, stream>>>(qb, kb, vb, ctxb);

  // 5) output projection + residual: x1 = x + ctx@wo + bo
  gemm_bf16_wmma<false,true,false,false,false,true,false>
      <<<gE, blk, 0, stream>>>(ctxb, wob, bo, x, nullptr, x1, nullptr,
                               NTOK, Ec, Ec, 1.0f);

  // 6) LN2 -> h2 (f32 for router) + h2b (bf16 for GEMMs)
  ln_kernel<<<NTOK, blk, 0, stream>>>(x1, ln2_g, ln2_b, h2, h2b);

  // 7) router + aux-loss stats
  zero_stats<<<1, 32, 0, stream>>>(stats);
  router_kernel<<<NTOK, blk, 0, stream>>>(h2, wgate, gates, stats, stats + 8);

  // 8) shared FFN: a1 = gelu(h2@sw1 + sb1); out = x1 + a1@sw2 + sb2
  gemm_bf16_wmma<true,false,false,false,false,false,true>
      <<<gD, blk, 0, stream>>>(h2b, sw1b, sb1, nullptr, nullptr, nullptr, a1b,
                               NTOK, DFFc, Ec, 1.0f);
  gemm_bf16_wmma<false,true,false,false,false,true,false>
      <<<gE, blk, 0, stream>>>(a1b, sw2b, sb2, x1, nullptr, outf, nullptr,
                               NTOK, Ec, DFFc, 1.0f);

  // 9) dense experts, gated accumulate into out
  for (int n = 0; n < NEc; ++n) {
    cvt_f32_bf16<<<1024, blk, 0, stream>>>(ew1 + (size_t)n * WED, ew1b, WED / 4);
    gemm_bf16_wmma<true,false,false,false,false,false,true>
        <<<gD, blk, 0, stream>>>(h2b, ew1b, eb1 + (size_t)n * DFFc, nullptr,
                                 nullptr, nullptr, eact, NTOK, DFFc, Ec, 1.0f);
    cvt_f32_bf16<<<1024, blk, 0, stream>>>(ew2 + (size_t)n * WED, ew2b, WED / 4);
    gemm_bf16_wmma<false,false,true,true,false,true,false>
        <<<gE, blk, 0, stream>>>(eact, ew2b, eb2 + (size_t)n * Ec, nullptr,
                                 gates + n, outf, nullptr, NTOK, Ec, DFFc, 1.0f);
  }

  // 10) aux loss scalar at the tail of d_out
  aux_kernel<<<1, 32, 0, stream>>>(stats, stats + 8, outf + NE_);
}
